// RWKV7_TimeMix_82411832475867
// MI455X (gfx1250) — matrix operand order repaired and checked by reading, not compiled, not run on hardware
//
#include <hip/hip_runtime.h>

#define C_DIM 1024
#define H_DIM 16
#define N_DIM 64
#define B_DIM 4
#define T_DIM 1024
#define M_DIM (B_DIM*T_DIM)   // 4096 rows
#define EPS_GN 0.00064f
#define BPAD 40               // padded LDS row stride (elements) for 32-K tile

typedef __attribute__((ext_vector_type(16))) __bf16 v16bf;
typedef __attribute__((ext_vector_type(8)))  __bf16 bf16x8;
typedef __attribute__((ext_vector_type(8)))  float  v8f;
typedef int v4i __attribute__((vector_size(16)));   // matches builtin param type

// --- gfx1250 async global->LDS path (guarded; falls back to sync staging) ---
#if defined(__has_builtin)
#  if __has_builtin(__builtin_amdgcn_global_load_async_to_lds_b128)
#    define HAVE_ASYNC_LDS 1
#  endif
#endif
#ifndef HAVE_ASYNC_LDS
#  define HAVE_ASYNC_LDS 0
#endif

#if defined(__has_builtin)
#  if __has_builtin(__builtin_amdgcn_s_wait_asynccnt)
#    define ASYNC_WAIT(n) __builtin_amdgcn_s_wait_asynccnt(n)
#  endif
#endif
#ifndef ASYNC_WAIT
#  define ASYNC_WAIT(n) asm volatile("s_wait_asynccnt %0" ::"i"(n) : "memory")
#endif

__device__ __forceinline__ float dev_sigmoid(float x) { return 1.0f / (1.0f + expf(-x)); }

// ---------------------------------------------------------------------------
// 1) token shift + channel mixes -> bf16
// ---------------------------------------------------------------------------
__global__ void mix_kernel(const float* __restrict__ x,
                           const float* __restrict__ m_r, const float* __restrict__ m_w,
                           const float* __restrict__ m_k, const float* __restrict__ m_v,
                           const float* __restrict__ m_a, const float* __restrict__ m_g,
                           __bf16* __restrict__ xr, __bf16* __restrict__ xw,
                           __bf16* __restrict__ xk, __bf16* __restrict__ xv,
                           __bf16* __restrict__ xa, __bf16* __restrict__ xg) {
  int idx = blockIdx.x * blockDim.x + threadIdx.x;
  if (idx >= M_DIM * C_DIM) return;
  int c = idx & (C_DIM - 1);
  int t = (idx / C_DIM) & (T_DIM - 1);
  float xv_ = x[idx];
  float prev = (t == 0) ? 0.0f : x[idx - C_DIM];
  float xx = prev - xv_;
  xr[idx] = (__bf16)(xv_ + xx * m_r[c]);
  xw[idx] = (__bf16)(xv_ + xx * m_w[c]);
  xk[idx] = (__bf16)(xv_ + xx * m_k[c]);
  xv[idx] = (__bf16)(xv_ + xx * m_v[c]);
  xa[idx] = (__bf16)(xv_ + xx * m_a[c]);
  xg[idx] = (__bf16)(xv_ + xx * m_g[c]);
}

// ---------------------------------------------------------------------------
// 2) weight transpose + convert: W[K,N] f32 -> Wt[N,K] bf16
// ---------------------------------------------------------------------------
__global__ void transpose_bf16_kernel(const float* __restrict__ W, __bf16* __restrict__ Wt,
                                      int K, int N) {
  int idx = blockIdx.x * blockDim.x + threadIdx.x;
  if (idx >= K * N) return;
  int k = idx / N, n = idx - k * N;
  Wt[(size_t)n * K + k] = (__bf16)W[idx];
}

// ---------------------------------------------------------------------------
// 3) bf16 WMMA GEMM: D[M,N] = epi(A[M,K] @ Bt[N,K]^T)
//    block tile 256x64 (8 waves, wave tile 32x64)
//    B tile (64x32) double-buffered in LDS, async-copied when available
//    EPI: 0=none 1=tanh 2=sigmoid 3=sigmoid(bias+x) 4=decay(bias,x)
// ---------------------------------------------------------------------------
template <int EPI, typename OutT>
__global__ void gemm_wmma_kernel(const __bf16* __restrict__ A, const __bf16* __restrict__ Bt,
                                 OutT* __restrict__ D, const float* __restrict__ bias,
                                 int M, int N, int K) {
  __shared__ __align__(16) __bf16 Bs[2][64 * BPAD];
  const int tid  = threadIdx.x;
  const int wave = tid >> 5;
  const int lane = tid & 31;
  const int half = lane >> 4;
  const int ln   = lane & 15;
  const int rowBase = blockIdx.y * 256 + wave * 32;
  const int colBase = blockIdx.x * 64;

  // staging map: 256 threads x 16B = 64 rows x 32 K of bf16
  const int srow = tid >> 2;  // 0..63
  const int sq   = tid & 3;   // 0..3
  const __bf16* bsrc = Bt + (size_t)(colBase + srow) * K + sq * 8;
  __bf16* bdst[2] = { &Bs[0][srow * BPAD + sq * 8], &Bs[1][srow * BPAD + sq * 8] };

  v8f acc[8] = {};

  const __bf16* Arow0 = A + (size_t)(rowBase + ln) * K;
  const __bf16* Arow1 = A + (size_t)(rowBase + 16 + ln) * K;

  auto stage = [&](int buf, int k0) {
#if HAVE_ASYNC_LDS
    __builtin_amdgcn_global_load_async_to_lds_b128(
        (__attribute__((address_space(1))) v4i*)(bsrc + k0),
        (__attribute__((address_space(3))) v4i*)bdst[buf], 0, 0);
#else
    *(bf16x8*)bdst[buf] = *(const bf16x8*)(bsrc + k0);
#endif
  };

  stage(0, 0);
  for (int k0 = 0; k0 < K; k0 += 32) {
    const int cur = (k0 >> 5) & 1;
    if (k0 + 32 < K) {
      stage(cur ^ 1, k0 + 32);
#if HAVE_ASYNC_LDS
      ASYNC_WAIT(1);   // oldest (current buffer) complete
#endif
    } else {
#if HAVE_ASYNC_LDS
      ASYNC_WAIT(0);
#endif
    }
    __syncthreads();

    // A fragments (direct from global): lane holds row, K {half*8+e} U {16+half*8+e}
    bf16x8 a0l = *(const bf16x8*)(Arow0 + k0 + half * 8);
    bf16x8 a0h = *(const bf16x8*)(Arow0 + k0 + 16 + half * 8);
    bf16x8 a1l = *(const bf16x8*)(Arow1 + k0 + half * 8);
    bf16x8 a1h = *(const bf16x8*)(Arow1 + k0 + 16 + half * 8);
    v16bf af0 = __builtin_shufflevector(a0l, a0h, 0,1,2,3,4,5,6,7,8,9,10,11,12,13,14,15);
    v16bf af1 = __builtin_shufflevector(a1l, a1h, 0,1,2,3,4,5,6,7,8,9,10,11,12,13,14,15);

    const __bf16* bb = &Bs[cur][0];
#pragma unroll
    for (int j = 0; j < 4; ++j) {
      // B fragment: lane holds col j*16+ln, K {half*8+e} U {16+half*8+e}
      const __bf16* bp = bb + (j * 16 + ln) * BPAD + half * 8;
      bf16x8 bl = *(const bf16x8*)bp;
      bf16x8 bh = *(const bf16x8*)(bp + 16);
      v16bf bf = __builtin_shufflevector(bl, bh, 0,1,2,3,4,5,6,7,8,9,10,11,12,13,14,15);
      acc[j]     = __builtin_amdgcn_wmma_f32_16x16x32_bf16(false, af0, false, bf, (short)0, acc[j],     false, false);
      acc[4 + j] = __builtin_amdgcn_wmma_f32_16x16x32_bf16(false, af1, false, bf, (short)0, acc[4 + j], false, false);
    }
    __syncthreads();
  }

  // C/D layout: VGPR r, lane<16 -> (M=r, N=lane); lane>=16 -> (M=r+8, N=lane-16)
#pragma unroll
  for (int s = 0; s < 2; ++s) {
    int rb = rowBase + s * 16;
#pragma unroll
    for (int r = 0; r < 8; ++r) {
      int row = rb + r + half * 8;
#pragma unroll
      for (int j = 0; j < 4; ++j) {
        int col = colBase + j * 16 + ln;
        float v = acc[s * 4 + j][r];
        float y;
        if constexpr (EPI == 0) {
          y = v;
        } else if constexpr (EPI == 1) {
          y = tanhf(v);
        } else if constexpr (EPI == 2) {
          y = dev_sigmoid(v);
        } else if constexpr (EPI == 3) {
          y = dev_sigmoid(bias[col] + v);
        } else {  // decay: w = -softplus(-(w0+v)) - 0.5 ; out = exp(-exp(w))
          float tv = bias[col] + v;
          float wv = -log1pf(expf(-tv)) - 0.5f;
          y = expf(-expf(wv));
        }
        D[(size_t)row * N + col] = (OutT)y;
      }
    }
  }
}

// ---------------------------------------------------------------------------
// 4) kk normalization + k' + scan-input prep (one block per (b,t,h))
// ---------------------------------------------------------------------------
__global__ void prep_kernel(const float* __restrict__ k, const float* __restrict__ a,
                            const float* __restrict__ k_k, const float* __restrict__ k_a,
                            float* __restrict__ kprime, float* __restrict__ akk,
                            float* __restrict__ bkk) {
  __shared__ float red[64];
  int i = threadIdx.x;
  int bh = blockIdx.x;
  int h = bh & (H_DIM - 1);
  int bt = bh >> 4;
  int c = h * N_DIM + i;
  size_t idx = (size_t)bt * C_DIM + c;
  float kv = k[idx], av = a[idx];
  float kkv = kv * k_k[c];
  red[i] = kkv * kkv;
  __syncthreads();
  for (int off = 32; off > 0; off >>= 1) {
    if (i < off) red[i] += red[i + off];
    __syncthreads();
  }
  float nrm = fmaxf(sqrtf(red[0]), 1e-12f);
  float kkn = kkv / nrm;
  kprime[idx] = kv * (1.0f + (av - 1.0f) * k_a[c]);
  akk[idx] = -kkn;
  bkk[idx] = kkn * av;
}

// ---------------------------------------------------------------------------
// 5) serial RWKV7 scan: one block (64 threads) per (b,h); S row in VGPRs
//    prefetch next timestep's operand vectors (global_prefetch_b8)
// ---------------------------------------------------------------------------
__global__ void scan_kernel(const float* __restrict__ r, const float* __restrict__ w,
                            const float* __restrict__ k, const float* __restrict__ ain,
                            const float* __restrict__ bin, const float* __restrict__ v,
                            float* __restrict__ o) {
  __shared__ float sr[64], sw[64], sk[64], sa[64], sb[64], sv[64];
  int i = threadIdx.x;
  int b = blockIdx.x / H_DIM;
  int h = blockIdx.x % H_DIM;
  float S[64];
#pragma unroll
  for (int j = 0; j < 64; ++j) S[j] = 0.0f;
  size_t base = (size_t)b * T_DIM * C_DIM + h * N_DIM;
  for (int t = 0; t < T_DIM; ++t) {
    size_t idx = base + (size_t)t * C_DIM;
    sr[i] = r[idx + i]; sw[i] = w[idx + i]; sk[i] = k[idx + i];
    sa[i] = ain[idx + i]; sb[i] = bin[idx + i]; sv[i] = v[idx + i];
    if (t + 1 < T_DIM) {
      size_t nx = idx + C_DIM + i;
      __builtin_prefetch(r + nx, 0, 1);
      __builtin_prefetch(w + nx, 0, 1);
      __builtin_prefetch(k + nx, 0, 1);
      __builtin_prefetch(ain + nx, 0, 1);
      __builtin_prefetch(bin + nx, 0, 1);
      __builtin_prefetch(v + nx, 0, 1);
    }
    __syncthreads();
    float saacc = 0.0f;
#pragma unroll
    for (int j = 0; j < 64; ++j) saacc += S[j] * sa[j];
    float vi = sv[i];
    float outv = 0.0f;
#pragma unroll
    for (int j = 0; j < 64; ++j) {
      S[j] = S[j] * sw[j] + saacc * sb[j] + vi * sk[j];
      outv += S[j] * sr[j];
    }
    o[idx + i] = outv;
    __syncthreads();
  }
}

// ---------------------------------------------------------------------------
// 6) group norm + r*k*r_k bonus + gate -> bf16 for final GEMM
// ---------------------------------------------------------------------------
__global__ void post_kernel(const float* __restrict__ o, const float* __restrict__ r,
                            const float* __restrict__ k, const float* __restrict__ v,
                            const float* __restrict__ r_k, const float* __restrict__ gamma,
                            const float* __restrict__ beta, const float* __restrict__ g,
                            __bf16* __restrict__ yg) {
  __shared__ float red[64];
  __shared__ float red2[64];
  int i = threadIdx.x;
  int bh = blockIdx.x;
  int h = bh & (H_DIM - 1);
  size_t bt = bh >> 4;
  int c = h * N_DIM + i;
  size_t idx = bt * C_DIM + c;
  float ov = o[idx];
  red[i] = ov; red2[i] = ov * ov;
  __syncthreads();
  for (int off = 32; off > 0; off >>= 1) {
    if (i < off) { red[i] += red[i + off]; red2[i] += red2[i + off]; }
    __syncthreads();
  }
  float mu = red[0] * (1.0f / 64.0f);
  float var = red2[0] * (1.0f / 64.0f) - mu * mu;
  float y = (ov - mu) * rsqrtf(var + EPS_GN) * gamma[c] + beta[c];
  __syncthreads();
  red[i] = r[idx] * k[idx] * r_k[c];
  __syncthreads();
  for (int off = 32; off > 0; off >>= 1) {
    if (i < off) red[i] += red[i + off];
    __syncthreads();
  }
  y += red[0] * v[idx];
  yg[idx] = (__bf16)(y * g[idx]);
}

// ---------------------------------------------------------------------------
extern "C" void kernel_launch(void* const* d_in, const int* in_sizes, int n_in,
                              void* d_out, int out_size, void* d_ws, size_t ws_size,
                              hipStream_t stream) {
  const float* x    = (const float*)d_in[0];
  const float* x_r  = (const float*)d_in[1];
  const float* x_w  = (const float*)d_in[2];
  const float* x_k  = (const float*)d_in[3];
  const float* x_v  = (const float*)d_in[4];
  const float* x_a  = (const float*)d_in[5];
  const float* x_g  = (const float*)d_in[6];
  const float* w0   = (const float*)d_in[7];
  const float* w1   = (const float*)d_in[8];
  const float* w2   = (const float*)d_in[9];
  const float* a0   = (const float*)d_in[10];
  const float* a1   = (const float*)d_in[11];
  const float* a2   = (const float*)d_in[12];
  // d_in[13..15] = v0,v1,v2 : unused in first-layer reference path
  const float* g1   = (const float*)d_in[16];
  const float* g2   = (const float*)d_in[17];
  const float* k_k  = (const float*)d_in[18];
  const float* k_a  = (const float*)d_in[19];
  const float* r_k  = (const float*)d_in[20];
  const float* Wr   = (const float*)d_in[21];
  const float* Wk   = (const float*)d_in[22];
  const float* Wv   = (const float*)d_in[23];
  const float* Wo   = (const float*)d_in[24];
  const float* lng  = (const float*)d_in[25];
  const float* lnb  = (const float*)d_in[26];

  float* outp   = (float*)d_out;
  float* vfirst = (float*)d_out + (size_t)M_DIM * C_DIM;  // second tuple output

  // workspace carving
  char* base = (char*)d_ws;
  size_t off = 0;
  auto carve = [&](size_t bytes) -> char* {
    off = (off + 255) & ~(size_t)255;
    char* p = base + off;
    off += bytes;
    return p;
  };
  const size_t MC = (size_t)M_DIM * C_DIM;
  const size_t CC = (size_t)C_DIM * C_DIM;

  __bf16* xr  = (__bf16*)carve(MC * 2);
  __bf16* xw  = (__bf16*)carve(MC * 2);
  __bf16* xk  = (__bf16*)carve(MC * 2);
  __bf16* xv  = (__bf16*)carve(MC * 2);
  __bf16* xa  = (__bf16*)carve(MC * 2);
  __bf16* xg  = (__bf16*)carve(MC * 2);
  __bf16* WrT = (__bf16*)carve(CC * 2);
  __bf16* WkT = (__bf16*)carve(CC * 2);
  __bf16* WvT = (__bf16*)carve(CC * 2);
  __bf16* WoT = (__bf16*)carve(CC * 2);
  __bf16* w1T = (__bf16*)carve((size_t)64 * C_DIM * 2);
  __bf16* w2T = (__bf16*)carve((size_t)C_DIM * 64 * 2);
  __bf16* a1T = (__bf16*)carve((size_t)64 * C_DIM * 2);
  __bf16* a2T = (__bf16*)carve((size_t)C_DIM * 64 * 2);
  __bf16* g1T = (__bf16*)carve((size_t)128 * C_DIM * 2);
  __bf16* g2T = (__bf16*)carve((size_t)C_DIM * 128 * 2);
  __bf16* hw  = (__bf16*)carve((size_t)M_DIM * 64 * 2);
  __bf16* ha  = (__bf16*)carve((size_t)M_DIM * 64 * 2);
  __bf16* hg  = (__bf16*)carve((size_t)M_DIM * 128 * 2);
  __bf16* yg  = (__bf16*)carve(MC * 2);
  float* rbuf   = (float*)carve(MC * 4);
  float* kraw   = (float*)carve(MC * 4);
  float* wdec   = (float*)carve(MC * 4);
  float* abuf   = (float*)carve(MC * 4);
  float* gbuf   = (float*)carve(MC * 4);
  float* kprime = (float*)carve(MC * 4);
  float* akkb   = (float*)carve(MC * 4);
  float* bkkb   = (float*)carve(MC * 4);
  float* obuf   = (float*)carve(MC * 4);
  (void)ws_size; (void)in_sizes; (void)n_in; (void)out_size;

  // 1) mixes
  mix_kernel<<<(int)((MC + 255) / 256), 256, 0, stream>>>(
      x, x_r, x_w, x_k, x_v, x_a, x_g, xr, xw, xk, xv, xa, xg);

  // 2) weight transposes
  transpose_bf16_kernel<<<(int)((CC + 255) / 256), 256, 0, stream>>>(Wr, WrT, C_DIM, C_DIM);
  transpose_bf16_kernel<<<(int)((CC + 255) / 256), 256, 0, stream>>>(Wk, WkT, C_DIM, C_DIM);
  transpose_bf16_kernel<<<(int)((CC + 255) / 256), 256, 0, stream>>>(Wv, WvT, C_DIM, C_DIM);
  transpose_bf16_kernel<<<(int)((CC + 255) / 256), 256, 0, stream>>>(Wo, WoT, C_DIM, C_DIM);
  transpose_bf16_kernel<<<(C_DIM * 64 + 255) / 256, 256, 0, stream>>>(w1, w1T, C_DIM, 64);
  transpose_bf16_kernel<<<(C_DIM * 64 + 255) / 256, 256, 0, stream>>>(w2, w2T, 64, C_DIM);
  transpose_bf16_kernel<<<(C_DIM * 64 + 255) / 256, 256, 0, stream>>>(a1, a1T, C_DIM, 64);
  transpose_bf16_kernel<<<(C_DIM * 64 + 255) / 256, 256, 0, stream>>>(a2, a2T, 64, C_DIM);
  transpose_bf16_kernel<<<(C_DIM * 128 + 255) / 256, 256, 0, stream>>>(g1, g1T, C_DIM, 128);
  transpose_bf16_kernel<<<(C_DIM * 128 + 255) / 256, 256, 0, stream>>>(g2, g2T, 128, C_DIM);

  dim3 blk(256);
  dim3 gridFull(C_DIM / 64, M_DIM / 256);   // N=1024
  dim3 gridN64(1, M_DIM / 256);             // N=64
  dim3 gridN128(2, M_DIM / 256);            // N=128

  // 3) GEMMs
  gemm_wmma_kernel<0, float><<<gridFull, blk, 0, stream>>>(xr, WrT, rbuf,  nullptr, M_DIM, C_DIM, C_DIM);
  gemm_wmma_kernel<0, float><<<gridFull, blk, 0, stream>>>(xk, WkT, kraw,  nullptr, M_DIM, C_DIM, C_DIM);
  gemm_wmma_kernel<0, float><<<gridFull, blk, 0, stream>>>(xv, WvT, vfirst, nullptr, M_DIM, C_DIM, C_DIM);
  gemm_wmma_kernel<1, __bf16><<<gridN64, blk, 0, stream>>>(xw, w1T, hw, nullptr, M_DIM, 64, C_DIM);
  gemm_wmma_kernel<4, float><<<gridFull, blk, 0, stream>>>(hw, w2T, wdec, w0, M_DIM, C_DIM, 64);
  gemm_wmma_kernel<0, __bf16><<<gridN64, blk, 0, stream>>>(xa, a1T, ha, nullptr, M_DIM, 64, C_DIM);
  gemm_wmma_kernel<3, float><<<gridFull, blk, 0, stream>>>(ha, a2T, abuf, a0, M_DIM, C_DIM, 64);
  gemm_wmma_kernel<2, __bf16><<<gridN128, blk, 0, stream>>>(xg, g1T, hg, nullptr, M_DIM, 128, C_DIM);
  gemm_wmma_kernel<0, float><<<gridFull, blk, 0, stream>>>(hg, g2T, gbuf, nullptr, M_DIM, C_DIM, 128);

  // 4) kk norm + scan inputs
  prep_kernel<<<M_DIM * H_DIM, 64, 0, stream>>>(kraw, abuf, k_k, k_a, kprime, akkb, bkkb);

  // 5) serial scan
  scan_kernel<<<B_DIM * H_DIM, 64, 0, stream>>>(rbuf, wdec, kprime, akkb, bkkb, vfirst, obuf);

  // 6) group norm + bonus + gate
  post_kernel<<<M_DIM * H_DIM, 64, 0, stream>>>(obuf, rbuf, kprime, vfirst, r_k, lng, lnb, gbuf, yg);

  // 7) output projection
  gemm_wmma_kernel<0, float><<<gridFull, blk, 0, stream>>>(yg, WoT, outp, nullptr, M_DIM, C_DIM, C_DIM);
}
